// RoutingCapsWithoutSquash_16904991277134
// MI455X (gfx1250) — matrix operand order repaired and checked by reading, not compiled, not run on hardware
//
#include <hip/hip_runtime.h>
#include <hip/hip_bf16.h>

typedef __attribute__((ext_vector_type(16))) __bf16         v16bf;
typedef __attribute__((ext_vector_type(16))) unsigned short v16u16;
typedef __attribute__((ext_vector_type(8)))  float          v8f;

#define B_SZ   256
#define N0_SZ  64
#define D0_SZ  128
#define N1_SZ  64
#define D1_SZ  128
#define BT     16            // batch tile (WMMA M)
#define NBT    (B_SZ / BT)   // 16 batch tiles

// ---------- scalar helpers ----------
static __device__ __forceinline__ unsigned short f32_to_bf16_bits(float f) {
    __bf16 b = (__bf16)f;                       // HW cvt (RNE) on gfx1250
    return __builtin_bit_cast(unsigned short, b);
}
static __device__ __forceinline__ float bf16_to_f32(unsigned short h) {
    union { unsigned u; float f; } c; c.u = ((unsigned)h) << 16;
    return c.f;
}
static __device__ __forceinline__ unsigned pack_bf16x2(float lo, float hi) {
    return ((unsigned)f32_to_bf16_bits(hi) << 16) | (unsigned)f32_to_bf16_bits(lo);
}

// =====================================================================
// FAST PATH: pre-converted bf16 fragments in d_ws
//
// x_frag layout (halfwords):  [bt][n0][kc(4)][lane(32)][i(16)]
//   lane = mrow + 16*khalf holds A row (bt*16+mrow);
//   element i is K = kc*32 + khalf*16 + i
// w_frag layout (halfwords):  [n1][n0][kc(4)][d1t(8)][lane(32)][i(16)]
//   lane holds B column d1t*16 + mrow; element i is K = kc*32 + khalf*16 + i
// => each lane's 16-element fragment is one contiguous 32B chunk.
// =====================================================================

__global__ __launch_bounds__(256) void convert_x_kernel(
    const float* __restrict__ x, unsigned short* __restrict__ xf)
{
    const int t    = blockIdx.x * 256 + threadIdx.x;    // [0, NBT*N0*4*32)
    const int lane = t & 31;
    const int kc   = (t >> 5) & 3;
    const int n0   = (t >> 7) & 63;
    const int bt   = t >> 13;
    const int mrow = lane & 15, khalf = lane >> 4;

    const float* src = x + ((size_t)(bt * BT + mrow) * N0_SZ + n0) * D0_SZ
                         + kc * 32 + khalf * 16;
    unsigned out[8];
#pragma unroll
    for (int j = 0; j < 4; ++j)          // i = 0..7   -> src[0..7]
        out[j] = pack_bf16x2(src[2 * j], src[2 * j + 1]);
#pragma unroll
    for (int j = 4; j < 8; ++j)          // i = 8..15  -> src[8..15]
        out[j] = pack_bf16x2(src[2 * j], src[2 * j + 1]);

    uint4* dst = (uint4*)(xf + (size_t)t * 16);
    dst[0] = make_uint4(out[0], out[1], out[2], out[3]);
    dst[1] = make_uint4(out[4], out[5], out[6], out[7]);
}

__global__ __launch_bounds__(256) void convert_w_kernel(
    const float* __restrict__ W, unsigned short* __restrict__ wf)
{
    const int t    = blockIdx.x * 256 + threadIdx.x;    // [0, N1*N0*4*8*32)
    const int lane = t & 31;
    const int d1t  = (t >> 5) & 7;
    const int kc   = (t >> 8) & 3;
    const int n0   = (t >> 10) & 63;
    const int n1   = t >> 16;
    const int mrow = lane & 15, khalf = lane >> 4;

    const float* src = W + (((size_t)n1 * N0_SZ + n0) * D0_SZ + kc * 32 + khalf * 16) * D1_SZ
                         + d1t * 16 + mrow;
    unsigned out[8];
#pragma unroll
    for (int j = 0; j < 8; ++j)          // element i at src[i*D1]
        out[j] = pack_bf16x2(src[(size_t)(2 * j) * D1_SZ], src[(size_t)(2 * j + 1) * D1_SZ]);

    uint4* dst = (uint4*)(wf + (size_t)t * 16);
    dst[0] = make_uint4(out[0], out[1], out[2], out[3]);
    dst[1] = make_uint4(out[4], out[5], out[6], out[7]);
}

static __device__ __forceinline__ v8f compute_u_tile_pre(
    const unsigned short* __restrict__ xf, const unsigned short* __restrict__ wf,
    int bt, int n1, int n0, int d1t, int lane)
{
    v8f acc = {0.f, 0.f, 0.f, 0.f, 0.f, 0.f, 0.f, 0.f};
    const unsigned short* xa = xf + ((((size_t)bt * N0_SZ + n0) * 4) * 32 + lane) * 16;
    const unsigned short* wb = wf + (((((size_t)n1 * N0_SZ + n0) * 4) * 8 + d1t) * 32 + lane) * 16;
#pragma unroll
    for (int kc = 0; kc < 4; ++kc) {
        v16u16 ab = *(const v16u16*)(xa + (size_t)kc * 32 * 16);
        v16u16 bb = *(const v16u16*)(wb + (size_t)kc * 8 * 32 * 16);
        acc = __builtin_amdgcn_wmma_f32_16x16x32_bf16(
            false, __builtin_bit_cast(v16bf, ab),
            false, __builtin_bit_cast(v16bf, bb),
            (short)0, acc, false, false);
    }
    return acc;
}

// Pass A (fast): u for all n0 -> LDS (bf16), s = sum_n0 u, logits -> d_ws.
__global__ __launch_bounds__(256) void pass_a_fast(
    const unsigned short* __restrict__ xf, const unsigned short* __restrict__ wf,
    float* __restrict__ logits)
{
    __shared__ unsigned short u_lds[N0_SZ][BT][D1_SZ]; // 256 KB
    __shared__ float          s_lds[BT][D1_SZ];        // 8 KB

    const int bt   = blockIdx.x;
    const int n1   = blockIdx.y;
    const int tid  = threadIdx.x;
    const int wave = tid >> 5;           // d1 tile
    const int lane = tid & 31;
    const int mrow = lane & 15, khalf = lane >> 4;
    const int d1base = wave * 16;

    float s_acc[8];
#pragma unroll
    for (int r = 0; r < 8; ++r) s_acc[r] = 0.f;

    for (int n0 = 0; n0 < N0_SZ; ++n0) {
        v8f acc = compute_u_tile_pre(xf, wf, bt, n1, n0, wave, lane);
#pragma unroll
        for (int r = 0; r < 8; ++r) {
            float v = acc[r];
            s_acc[r] += v;
            u_lds[n0][khalf * 8 + r][d1base + mrow] = f32_to_bf16_bits(v);
        }
    }
#pragma unroll
    for (int r = 0; r < 8; ++r)
        s_lds[khalf * 8 + r][d1base + mrow] = s_acc[r];
    __syncthreads();

    const float inv_sqrt_d1 = 0.08838834764831844f; // 1/sqrt(128)
    for (int idx = tid; idx < BT * N0_SZ; idx += 256) {
        const int bl = idx >> 6;
        const int n0 = idx & (N0_SZ - 1);
        float d = 0.f;
#pragma unroll 4
        for (int k = 0; k < D1_SZ; ++k)
            d += bf16_to_f32(u_lds[n0][bl][k]) * s_lds[bl][k];
        logits[((size_t)(bt * BT + bl) * N1_SZ + n1) * N0_SZ + n0] = d * inv_sqrt_d1;
    }
}

// Pass C (fast): recompute u, out[b,n1,:] = sum_n0 u * c.
__global__ __launch_bounds__(256) void pass_c_fast(
    const unsigned short* __restrict__ xf, const unsigned short* __restrict__ wf,
    const float* __restrict__ probs, float* __restrict__ out)
{
    __shared__ float p_lds[BT][N0_SZ];

    const int bt   = blockIdx.x;
    const int n1   = blockIdx.y;
    const int tid  = threadIdx.x;
    const int wave = tid >> 5;
    const int lane = tid & 31;
    const int mrow = lane & 15, khalf = lane >> 4;
    const int d1base = wave * 16;

    for (int idx = tid; idx < BT * N0_SZ; idx += 256) {
        const int bl = idx >> 6;
        const int n0 = idx & (N0_SZ - 1);
        p_lds[bl][n0] = probs[((size_t)(bt * BT + bl) * N1_SZ + n1) * N0_SZ + n0];
    }
    __syncthreads();

    float oacc[8];
#pragma unroll
    for (int r = 0; r < 8; ++r) oacc[r] = 0.f;

    for (int n0 = 0; n0 < N0_SZ; ++n0) {
        v8f acc = compute_u_tile_pre(xf, wf, bt, n1, n0, wave, lane);
#pragma unroll
        for (int r = 0; r < 8; ++r)
            oacc[r] += acc[r] * p_lds[khalf * 8 + r][n0];
    }
#pragma unroll
    for (int r = 0; r < 8; ++r)
        out[((size_t)(bt * BT + khalf * 8 + r) * N1_SZ + n1) * D1_SZ + d1base + mrow] = oacc[r];
}

// Pass B: softmax over n1 per (b,n0) row, then += bias.
__global__ __launch_bounds__(256) void softmax_kernel(
    float* __restrict__ lg, const float* __restrict__ bias)
{
    const int idx = blockIdx.x * blockDim.x + threadIdx.x;
    const int b   = idx >> 6;
    const int n0  = idx & (N0_SZ - 1);
    float v[N1_SZ];
    float m = -3.4e38f;
#pragma unroll 8
    for (int n1 = 0; n1 < N1_SZ; ++n1) {
        v[n1] = lg[((size_t)b * N1_SZ + n1) * N0_SZ + n0];
        m = fmaxf(m, v[n1]);
    }
    float s = 0.f;
#pragma unroll 8
    for (int n1 = 0; n1 < N1_SZ; ++n1) { v[n1] = __expf(v[n1] - m); s += v[n1]; }
    const float inv = 1.f / s;
#pragma unroll 8
    for (int n1 = 0; n1 < N1_SZ; ++n1)
        lg[((size_t)b * N1_SZ + n1) * N0_SZ + n0] = v[n1] * inv + bias[n1 * N0_SZ + n0];
}

// =====================================================================
// SLOW PATH (fallback if d_ws too small): convert inside the GEMM loop
// =====================================================================
static __device__ __forceinline__ v8f compute_u_tile_slow(
    const float* __restrict__ x, const float* __restrict__ W,
    int b0, int n1, int n0, int d1base, int lane)
{
    const int mrow = lane & 15, khalf = lane >> 4;
    v8f acc = {0.f, 0.f, 0.f, 0.f, 0.f, 0.f, 0.f, 0.f};
    const float* xbase = x + ((size_t)(b0 + mrow) * N0_SZ + n0) * D0_SZ + khalf * 16;
    const float* wbase = W + (((size_t)n1 * N0_SZ + n0) * D0_SZ + khalf * 16) * D1_SZ
                           + d1base + mrow;
#pragma unroll
    for (int kc = 0; kc < D0_SZ; kc += 32) {
        v16u16 ab, bb;
#pragma unroll
        for (int j = 0; j < 8; ++j) {
            ab[j]     = f32_to_bf16_bits(xbase[kc + j]);
            ab[8 + j] = f32_to_bf16_bits(xbase[kc + 8 + j]);
        }
#pragma unroll
        for (int i = 0; i < 16; ++i)
            bb[i] = f32_to_bf16_bits(wbase[(size_t)(kc + i) * D1_SZ]);
        acc = __builtin_amdgcn_wmma_f32_16x16x32_bf16(
            false, __builtin_bit_cast(v16bf, ab),
            false, __builtin_bit_cast(v16bf, bb),
            (short)0, acc, false, false);
    }
    return acc;
}

__global__ __launch_bounds__(256) void pass_a_slow(
    const float* __restrict__ x, const float* __restrict__ W, float* __restrict__ logits)
{
    __shared__ unsigned short u_lds[N0_SZ][BT][D1_SZ];
    __shared__ float          s_lds[BT][D1_SZ];
    const int b0 = blockIdx.x * BT, n1 = blockIdx.y, tid = threadIdx.x;
    const int wave = tid >> 5, lane = tid & 31;
    const int mrow = lane & 15, khalf = lane >> 4, d1base = wave * 16;
    float s_acc[8];
#pragma unroll
    for (int r = 0; r < 8; ++r) s_acc[r] = 0.f;
    for (int n0 = 0; n0 < N0_SZ; ++n0) {
        v8f acc = compute_u_tile_slow(x, W, b0, n1, n0, d1base, lane);
#pragma unroll
        for (int r = 0; r < 8; ++r) {
            float v = acc[r];
            s_acc[r] += v;
            u_lds[n0][khalf * 8 + r][d1base + mrow] = f32_to_bf16_bits(v);
        }
    }
#pragma unroll
    for (int r = 0; r < 8; ++r) s_lds[khalf * 8 + r][d1base + mrow] = s_acc[r];
    __syncthreads();
    const float inv_sqrt_d1 = 0.08838834764831844f;
    for (int idx = tid; idx < BT * N0_SZ; idx += 256) {
        const int bl = idx >> 6, n0 = idx & (N0_SZ - 1);
        float d = 0.f;
#pragma unroll 4
        for (int k = 0; k < D1_SZ; ++k)
            d += bf16_to_f32(u_lds[n0][bl][k]) * s_lds[bl][k];
        logits[((size_t)(b0 + bl) * N1_SZ + n1) * N0_SZ + n0] = d * inv_sqrt_d1;
    }
}

__global__ __launch_bounds__(256) void pass_c_slow(
    const float* __restrict__ x, const float* __restrict__ W,
    const float* __restrict__ probs, float* __restrict__ out)
{
    __shared__ float p_lds[BT][N0_SZ];
    const int b0 = blockIdx.x * BT, n1 = blockIdx.y, tid = threadIdx.x;
    const int wave = tid >> 5, lane = tid & 31;
    const int mrow = lane & 15, khalf = lane >> 4, d1base = wave * 16;
    for (int idx = tid; idx < BT * N0_SZ; idx += 256) {
        const int bl = idx >> 6, n0 = idx & (N0_SZ - 1);
        p_lds[bl][n0] = probs[((size_t)(b0 + bl) * N1_SZ + n1) * N0_SZ + n0];
    }
    __syncthreads();
    float oacc[8];
#pragma unroll
    for (int r = 0; r < 8; ++r) oacc[r] = 0.f;
    for (int n0 = 0; n0 < N0_SZ; ++n0) {
        v8f acc = compute_u_tile_slow(x, W, b0, n1, n0, d1base, lane);
#pragma unroll
        for (int r = 0; r < 8; ++r)
            oacc[r] += acc[r] * p_lds[khalf * 8 + r][n0];
    }
#pragma unroll
    for (int r = 0; r < 8; ++r)
        out[((size_t)(b0 + khalf * 8 + r) * N1_SZ + n1) * D1_SZ + d1base + mrow] = oacc[r];
}

// =====================================================================
extern "C" void kernel_launch(void* const* d_in, const int* in_sizes, int n_in,
                              void* d_out, int out_size, void* d_ws, size_t ws_size,
                              hipStream_t stream) {
    const float* x    = (const float*)d_in[0];   // [B, N0, D0]
    const float* W    = (const float*)d_in[1];   // [N1, N0, D0, D1]
    const float* bias = (const float*)d_in[2];   // [N1, N0, 1]
    float* out = (float*)d_out;                  // [B, N1, D1]

    const size_t logits_bytes = (size_t)B_SZ * N1_SZ * N0_SZ * 4;          // 4 MB
    const size_t xf_bytes     = (size_t)B_SZ * N0_SZ * D0_SZ * 2;          // 4 MB
    const size_t wf_bytes     = (size_t)N1_SZ * N0_SZ * D0_SZ * D1_SZ * 2; // 128 MB

    float* logits = (float*)d_ws;
    dim3 grid(NBT, N1_SZ);

    if (ws_size >= logits_bytes + xf_bytes + wf_bytes) {
        unsigned short* xf = (unsigned short*)((char*)d_ws + logits_bytes);
        unsigned short* wf = (unsigned short*)((char*)d_ws + logits_bytes + xf_bytes);
        convert_x_kernel<<<(NBT * N0_SZ * 4 * 32) / 256, 256, 0, stream>>>(x, xf);
        convert_w_kernel<<<(N1_SZ * N0_SZ * 4 * 8 * 32) / 256, 256, 0, stream>>>(W, wf);
        pass_a_fast<<<grid, 256, 0, stream>>>(xf, wf, logits);
        softmax_kernel<<<(B_SZ * N0_SZ) / 256, 256, 0, stream>>>(logits, bias);
        pass_c_fast<<<grid, 256, 0, stream>>>(xf, wf, logits, out);
    } else {
        pass_a_slow<<<grid, 256, 0, stream>>>(x, W, logits);
        softmax_kernel<<<(B_SZ * N0_SZ) / 256, 256, 0, stream>>>(logits, bias);
        pass_c_slow<<<grid, 256, 0, stream>>>(x, W, logits, out);
    }
}
